// CausalSelfAttention_32057635898092
// MI455X (gfx1250) — matrix operand order repaired and checked by reading, not compiled, not run on hardware
//
#include <hip/hip_runtime.h>
#include <hip/hip_bf16.h>
#include <stddef.h>

// ---------------------------------------------------------------------------
// Causal self-attention, B=4 T=2048 D=1024 H=16 DK=64, fp32 in/out.
// Strategy (MI455X / gfx1250, wave32, WMMA):
//   fp32 -> bf16 once, all GEMMs + attention via v_wmma_f32_16x16x32_bf16,
//   flash-attention (online softmax) so the 2048x2048 score matrix never
//   touches HBM.  ~137 GFLOP total, ~hundreds MB of traffic -> matrix-core
//   bound; bf16 WMMA with f32 accumulate is the throughput path.
// ---------------------------------------------------------------------------

#define B_  4
#define T_  2048
#define D_  1024
#define H_  16
#define DK_ 64
#define BT_ (B_ * T_)          // 8192 rows

typedef __attribute__((ext_vector_type(16))) __bf16 v16bf;
typedef __attribute__((ext_vector_type(8)))  __bf16 v8bf;
typedef __attribute__((ext_vector_type(8)))  float  v8f;

static __device__ __forceinline__ v16bf cat8(v8bf lo, v8bf hi) {
  return __builtin_shufflevector(lo, hi, 0,1,2,3,4,5,6,7,8,9,10,11,12,13,14,15);
}

static __device__ __forceinline__ v8f wmma_bf16(v16bf a, v16bf b, v8f c) {
  // (neg_a, A, neg_b, B, c_mod, C, reuse_a, reuse_b)
  return __builtin_amdgcn_wmma_f32_16x16x32_bf16(false, a, false, b, (short)0, c,
                                                 false, false);
}

// -------------------------- conversion kernels ----------------------------

__global__ void csa_cvt_x(const float* __restrict__ x, __bf16* __restrict__ xb, int n) {
  int idx = blockIdx.x * blockDim.x + threadIdx.x;
  if (idx < n) xb[idx] = (__bf16)x[idx];
}

// Wt[n*D + k] = W[k*D + n]  (transpose so WMMA B-fragments are contiguous in K)
__global__ void csa_cvt_wT(const float* __restrict__ Wq, const float* __restrict__ Wk,
                           const float* __restrict__ Wv, const float* __restrict__ Wo,
                           __bf16* __restrict__ WqT, __bf16* __restrict__ WkT,
                           __bf16* __restrict__ WvT, __bf16* __restrict__ WoT) {
  const float* src; __bf16* dst;
  switch (blockIdx.y) {
    case 0:  src = Wq; dst = WqT; break;
    case 1:  src = Wk; dst = WkT; break;
    case 2:  src = Wv; dst = WvT; break;
    default: src = Wo; dst = WoT; break;
  }
  int idx = blockIdx.x * blockDim.x + threadIdx.x;   // over D*D, write-coalesced
  int n = idx / D_;
  int kk = idx - n * D_;
  dst[idx] = (__bf16)src[kk * D_ + n];
}

// ----------------------------- QKV projection -----------------------------
// C[8192x1024] = Xbf16 @ W + bias, block tile 128x64 (8 waves, wave = 32x32).
// which==0 -> q (scaled 1/sqrt(DK)) stored [b][h][t][dk]
// which==1 -> k stored [b][h][t][dk]
// which==2 -> v stored transposed [b][h][dk][t]
__global__ void csa_qkv_gemm(const __bf16* __restrict__ xb,
                             const __bf16* __restrict__ WqT,
                             const __bf16* __restrict__ WkT,
                             const __bf16* __restrict__ WvT,
                             const float* __restrict__ bq,
                             const float* __restrict__ bk,
                             const float* __restrict__ bv,
                             __bf16* __restrict__ qout,
                             __bf16* __restrict__ kout,
                             __bf16* __restrict__ vTout) {
  const int which = blockIdx.z;
  const __bf16* Wt   = (which == 0) ? WqT : (which == 1) ? WkT : WvT;
  const float*  bias = (which == 0) ? bq  : (which == 1) ? bk  : bv;

  const int lane = threadIdx.x & 31;
  const int wid  = threadIdx.x >> 5;
  const int wm = wid & 3, wn = wid >> 2;
  const int m0 = blockIdx.x * 128 + wm * 32;
  const int n0 = blockIdx.y * 64  + wn * 32;
  const int lr = lane & 15;
  const int hi = lane >> 4;

  v8f acc[2][2] = {};
  for (int k0 = 0; k0 < D_; k0 += 32) {
    v16bf a[2], bb[2];
#pragma unroll
    for (int mi = 0; mi < 2; ++mi) {
      const __bf16* p = xb + (size_t)(m0 + 16 * mi + lr) * D_ + k0 + hi * 8;
      a[mi] = cat8(*(const v8bf*)p, *(const v8bf*)(p + 16));
    }
#pragma unroll
    for (int ni = 0; ni < 2; ++ni) {
      const __bf16* p = Wt + (size_t)(n0 + 16 * ni + lr) * D_ + k0 + hi * 8;
      bb[ni] = cat8(*(const v8bf*)p, *(const v8bf*)(p + 16));
    }
#pragma unroll
    for (int mi = 0; mi < 2; ++mi)
#pragma unroll
      for (int ni = 0; ni < 2; ++ni)
        acc[mi][ni] = wmma_bf16(a[mi], bb[ni], acc[mi][ni]);
  }

#pragma unroll
  for (int mi = 0; mi < 2; ++mi) {
#pragma unroll
    for (int ni = 0; ni < 2; ++ni) {
      const int col  = n0 + 16 * ni + lr;
      const float bv_ = bias[col];
      const int hh = col >> 6;      // head    (DK = 64)
      const int dk = col & 63;
#pragma unroll
      for (int r = 0; r < 8; ++r) {
        const int row  = m0 + 16 * mi + r + 8 * hi;
        const int bidx = row >> 11;   // / T_
        const int t    = row & (T_ - 1);
        float v = acc[mi][ni][r] + bv_;
        if (which == 0) {
          v *= 0.125f;  // 1/sqrt(64) folded into q
          qout[(size_t)((bidx * H_ + hh) * T_ + t) * DK_ + dk] = (__bf16)v;
        } else if (which == 1) {
          kout[(size_t)((bidx * H_ + hh) * T_ + t) * DK_ + dk] = (__bf16)v;
        } else {
          vTout[(size_t)((bidx * H_ + hh) * DK_ + dk) * T_ + t] = (__bf16)v;
        }
      }
    }
  }
}

// ----------------------------- flash attention ----------------------------
// grid = (T/128, B*H); 8 waves, each wave owns 16 query rows.
// Streams K/V in 32-key tiles: QK^T = 4 wmma, online softmax with shfl_xor
// reductions (stay inside 16-lane halves), P re-laid out through LDS
// (C-tile -> A-fragment), PV = 4 wmma.
__global__ void csa_flash_attn(const __bf16* __restrict__ q,
                               const __bf16* __restrict__ k,
                               const __bf16* __restrict__ vT,
                               __bf16* __restrict__ attnb) {
  __shared__ __bf16 plds[8][16][32];   // per-wave P staging (8 KB)

  const int lane = threadIdx.x & 31;
  const int wid  = threadIdx.x >> 5;
  const int lr = lane & 15;
  const int hi = lane >> 4;
  const int bh   = blockIdx.y;
  const int bidx = bh >> 4;    // / H_
  const int hh   = bh & 15;
  const int qbase = blockIdx.x * 128 + wid * 16;

  const __bf16* qh = q  + (size_t)bh * T_ * DK_;
  const __bf16* kh = k  + (size_t)bh * T_ * DK_;
  const __bf16* vh = vT + (size_t)bh * DK_ * T_;

  // Q fragments (16 rows x 64 dk -> two K=32 A-fragments), loaded once
  v16bf qa[2];
#pragma unroll
  for (int d = 0; d < 2; ++d) {
    const __bf16* p = qh + (size_t)(qbase + lr) * DK_ + d * 32 + hi * 8;
    qa[d] = cat8(*(const v8bf*)p, *(const v8bf*)(p + 16));
  }

  float mrow[8], lrow[8];
  v8f o[4] = {};
#pragma unroll
  for (int r = 0; r < 8; ++r) { mrow[r] = -1e30f; lrow[r] = 0.f; }

  const int kend = blockIdx.x * 128 + 128;   // block-uniform trip count
  for (int kt = 0; kt < kend; kt += 32) {
    v8f s[2] = {};
#pragma unroll
    for (int si = 0; si < 2; ++si) {
      const int ktile = kt + si * 16;
#pragma unroll
      for (int d = 0; d < 2; ++d) {
        const __bf16* p = kh + (size_t)(ktile + lr) * DK_ + d * 32 + hi * 8;
        v16bf kb = cat8(*(const v8bf*)p, *(const v8bf*)(p + 16));
        s[si] = wmma_bf16(qa[d], kb, s[si]);
      }
#pragma unroll
      for (int r = 0; r < 8; ++r) {
        const int qi = qbase + r + 8 * hi;
        if (ktile + lr > qi) s[si][r] = -1e30f;   // finite causal mask
      }
    }

    // online softmax, per row (row r+8*hi lives in a 16-lane half)
#pragma unroll
    for (int r = 0; r < 8; ++r) {
      float t = fmaxf(s[0][r], s[1][r]);
      t = fmaxf(t, __shfl_xor(t, 1, 32));
      t = fmaxf(t, __shfl_xor(t, 2, 32));
      t = fmaxf(t, __shfl_xor(t, 4, 32));
      t = fmaxf(t, __shfl_xor(t, 8, 32));
      const float mnew  = fmaxf(mrow[r], t);
      const float alpha = __expf(mrow[r] - mnew);
      const float p0 = __expf(s[0][r] - mnew);
      const float p1 = __expf(s[1][r] - mnew);
      float rs = p0 + p1;
      rs += __shfl_xor(rs, 1, 32);
      rs += __shfl_xor(rs, 2, 32);
      rs += __shfl_xor(rs, 4, 32);
      rs += __shfl_xor(rs, 8, 32);
      lrow[r] = lrow[r] * alpha + rs;
      mrow[r] = mnew;
#pragma unroll
      for (int dt = 0; dt < 4; ++dt) o[dt][r] *= alpha;
      plds[wid][r + 8 * hi][lr]      = (__bf16)p0;
      plds[wid][r + 8 * hi][lr + 16] = (__bf16)p1;
    }
    __syncthreads();

    // reload P as an A-fragment (16 q-rows x 32 keys)
    v16bf pa;
    {
      const __bf16* p = &plds[wid][lr][hi * 8];
      pa = cat8(*(const v8bf*)p, *(const v8bf*)(p + 16));
    }
    // PV: V^T stored [dk][t] -> contiguous B-fragment loads
#pragma unroll
    for (int dt = 0; dt < 4; ++dt) {
      const __bf16* p = vh + (size_t)(dt * 16 + lr) * T_ + kt + hi * 8;
      v16bf vb = cat8(*(const v8bf*)p, *(const v8bf*)(p + 16));
      o[dt] = wmma_bf16(pa, vb, o[dt]);
    }
    __syncthreads();
  }

  // normalize and write [b][t][h*DK+dk] (row-major rows for the final GEMM)
#pragma unroll
  for (int dt = 0; dt < 4; ++dt) {
#pragma unroll
    for (int r = 0; r < 8; ++r) {
      const int t   = qbase + r + 8 * hi;
      const int col = hh * DK_ + dt * 16 + lr;
      const float v = o[dt][r] / lrow[r];
      attnb[(size_t)(bidx * T_ + t) * D_ + col] = (__bf16)v;
    }
  }
}

// --------------------------- output projection ----------------------------
__global__ void csa_out_gemm(const __bf16* __restrict__ attnb,
                             const __bf16* __restrict__ WoT,
                             const float* __restrict__ bo,
                             float* __restrict__ out) {
  const int lane = threadIdx.x & 31;
  const int wid  = threadIdx.x >> 5;
  const int wm = wid & 3, wn = wid >> 2;
  const int m0 = blockIdx.x * 128 + wm * 32;
  const int n0 = blockIdx.y * 64  + wn * 32;
  const int lr = lane & 15;
  const int hi = lane >> 4;

  v8f acc[2][2] = {};
  for (int k0 = 0; k0 < D_; k0 += 32) {
    v16bf a[2], bb[2];
#pragma unroll
    for (int mi = 0; mi < 2; ++mi) {
      const __bf16* p = attnb + (size_t)(m0 + 16 * mi + lr) * D_ + k0 + hi * 8;
      a[mi] = cat8(*(const v8bf*)p, *(const v8bf*)(p + 16));
    }
#pragma unroll
    for (int ni = 0; ni < 2; ++ni) {
      const __bf16* p = WoT + (size_t)(n0 + 16 * ni + lr) * D_ + k0 + hi * 8;
      bb[ni] = cat8(*(const v8bf*)p, *(const v8bf*)(p + 16));
    }
#pragma unroll
    for (int mi = 0; mi < 2; ++mi)
#pragma unroll
      for (int ni = 0; ni < 2; ++ni)
        acc[mi][ni] = wmma_bf16(a[mi], bb[ni], acc[mi][ni]);
  }

#pragma unroll
  for (int mi = 0; mi < 2; ++mi) {
#pragma unroll
    for (int ni = 0; ni < 2; ++ni) {
      const int col = n0 + 16 * ni + lr;
      const float bv_ = bo[col];
#pragma unroll
      for (int r = 0; r < 8; ++r) {
        const int row = m0 + 16 * mi + r + 8 * hi;
        out[(size_t)row * D_ + col] = acc[mi][ni][r] + bv_;
      }
    }
  }
}

// ------------------------------- launcher ---------------------------------

extern "C" void kernel_launch(void* const* d_in, const int* in_sizes, int n_in,
                              void* d_out, int out_size, void* d_ws, size_t ws_size,
                              hipStream_t stream) {
  const float* x  = (const float*)d_in[0];
  const float* Wq = (const float*)d_in[1];
  const float* bq = (const float*)d_in[2];
  const float* Wk = (const float*)d_in[3];
  const float* bk = (const float*)d_in[4];
  const float* Wv = (const float*)d_in[5];
  const float* bv = (const float*)d_in[6];
  const float* Wo = (const float*)d_in[7];
  const float* bo = (const float*)d_in[8];

  // carve bf16 workspace
  char* ws = (char*)d_ws;
  size_t off = 0;
  auto carve = [&](size_t bytes) -> void* {
    void* p = ws + off;
    off += (bytes + 255) & ~(size_t)255;
    return p;
  };
  __bf16* xb    = (__bf16*)carve((size_t)BT_ * D_ * 2);
  __bf16* WqT   = (__bf16*)carve((size_t)D_ * D_ * 2);
  __bf16* WkT   = (__bf16*)carve((size_t)D_ * D_ * 2);
  __bf16* WvT   = (__bf16*)carve((size_t)D_ * D_ * 2);
  __bf16* WoT   = (__bf16*)carve((size_t)D_ * D_ * 2);
  __bf16* qb    = (__bf16*)carve((size_t)BT_ * D_ * 2);
  __bf16* kb    = (__bf16*)carve((size_t)BT_ * D_ * 2);
  __bf16* vTb   = (__bf16*)carve((size_t)BT_ * D_ * 2);
  __bf16* attnb = (__bf16*)carve((size_t)BT_ * D_ * 2);

  const int nx = B_ * T_ * D_;
  csa_cvt_x<<<(nx + 255) / 256, 256, 0, stream>>>(x, xb, nx);
  csa_cvt_wT<<<dim3(D_ * D_ / 256, 4), 256, 0, stream>>>(Wq, Wk, Wv, Wo,
                                                         WqT, WkT, WvT, WoT);
  csa_qkv_gemm<<<dim3(BT_ / 128, D_ / 64, 3), 256, 0, stream>>>(
      xb, WqT, WkT, WvT, bq, bk, bv, qb, kb, vTb);
  csa_flash_attn<<<dim3(T_ / 128, B_ * H_), 256, 0, stream>>>(qb, kb, vTb, attnb);
  csa_out_gemm<<<dim3(BT_ / 128, D_ / 64), 256, 0, stream>>>(attnb, WoT, bo,
                                                             (float*)d_out);
}
